// box_tower_large_69681549410530
// MI455X (gfx1250) — compile-verified, matrix operand order checked by reading, not run on hardware
//
#include <hip/hip_runtime.h>

// ---------------------------------------------------------------------------
// CDNA5 (gfx1250) implementation: f16 WMMA implicit-GEMM conv tower pipeline.
// wave32; v_wmma_f32_16x16x32_f16 for all GEMM math; TDM (tensor_load_to_lds)
// stages L2-resident weight tiles; NHWC f16 intermediates for b128 traffic.
// ---------------------------------------------------------------------------

typedef _Float16 h16;
typedef __attribute__((ext_vector_type(16))) _Float16 v16h;
typedef __attribute__((ext_vector_type(8)))  _Float16 v8h;
typedef __attribute__((ext_vector_type(8)))  float    v8f;
typedef __attribute__((ext_vector_type(4)))  unsigned int u32x4;
typedef __attribute__((ext_vector_type(8)))  int          i32x8;
typedef __attribute__((ext_vector_type(4)))  int          i32x4;

union FragAB { v16h v; v8h h[2]; };

__device__ __forceinline__ v8f wmma16x16x32(v16h a, v16h b, v8f c) {
  // (neg_a, A, neg_b, B, c_mod, C, reuse_a, reuse_b)
  return __builtin_amdgcn_wmma_f32_16x16x32_f16(false, a, false, b, (short)0, c,
                                                false, false);
}

// ===========================================================================
// Kernel 1: BN(z), BN(x) fused + pixelwise correlation  corr = z^T x
//   z: (112x64) per batch, x: (112x1024) per batch.
//   Output corr in NHWC-f16: [b][pixel][64 chan]  (chan innermost).
//   K padded 112 -> 128 with zeros. Block: one batch x 128-pixel chunk.
// ===========================================================================
__global__ __launch_bounds__(256) void corr_wmma(
    const float* __restrict__ search, const float* __restrict__ kern,
    const float* __restrict__ zg, const float* __restrict__ zbb,
    const float* __restrict__ zm, const float* __restrict__ zv,
    const float* __restrict__ xg, const float* __restrict__ xbb,
    const float* __restrict__ xm, const float* __restrict__ xv,
    h16* __restrict__ corr)
{
  const int b   = blockIdx.y;
  const int px0 = blockIdx.x * 128;
  const int tid = threadIdx.x;
  const int wv = tid >> 5, lane = tid & 31, lr = lane & 15, hi = lane >> 4;

  __shared__ float scz[128], shz[128], scx[128], shx[128];
  __shared__ __align__(16) h16 lZ[64 * 136];    // [kpix][cin pad 136] f16
  __shared__ __align__(16) h16 lX[128 * 136];   // [pix ][cin pad 136] f16

  if (tid < 128) {
    float s = 0.f, sh = 0.f;
    if (tid < 112) { s = zg[tid] * rsqrtf(zv[tid] + 1e-5f); sh = zbb[tid] - zm[tid] * s; }
    scz[tid] = s; shz[tid] = sh;
  } else {
    int c = tid - 128;
    float s = 0.f, sh = 0.f;
    if (c < 112) { s = xg[c] * rsqrtf(xv[c] + 1e-5f); sh = xbb[c] - xm[c] * s; }
    scx[c] = s; shx[c] = sh;
  }
  __syncthreads();

  // stage z^T (BN applied), zeros for c in [112,128)
  for (int i = tid; i < 64 * 128; i += 256) {
    int kp = i & 63, c = i >> 6;
    float v = (c < 112) ? kern[((size_t)(b * 112 + c)) * 64 + kp] * scz[c] + shz[c] : 0.f;
    lZ[kp * 136 + c] = (h16)v;
  }
  // stage x chunk (BN applied)
  for (int i = tid; i < 128 * 128; i += 256) {
    int px = i & 127, c = i >> 7;
    float v = (c < 112) ? search[((size_t)(b * 112 + c)) * 1024 + px0 + px] * scx[c] + shx[c] : 0.f;
    lX[px * 136 + c] = (h16)v;
  }
  __syncthreads();

  const int wm = wv & 1;        // M-tile pair: {0,1} or {2,3}
  const int wn = wv >> 1;       // N-tile pair: 0..3
  const int k0off = hi ? 8 : 0; // A/B lane-half K split per ISA layout
  v8f acc[2][2] = {};

#pragma unroll
  for (int kc = 0; kc < 4; ++kc) {
    const int kb = kc * 32 + k0off;
    FragAB af[2], bf[2];
#pragma unroll
    for (int mi = 0; mi < 2; ++mi) {
      int kp = (2 * wm + mi) * 16 + lr;
      af[mi].h[0] = *(const v8h*)&lZ[kp * 136 + kb];
      af[mi].h[1] = *(const v8h*)&lZ[kp * 136 + kb + 16];
    }
#pragma unroll
    for (int ni = 0; ni < 2; ++ni) {
      int px = (2 * wn + ni) * 16 + lr;
      bf[ni].h[0] = *(const v8h*)&lX[px * 136 + kb];
      bf[ni].h[1] = *(const v8h*)&lX[px * 136 + kb + 16];
    }
    acc[0][0] = wmma16x16x32(af[0].v, bf[0].v, acc[0][0]);
    acc[0][1] = wmma16x16x32(af[0].v, bf[1].v, acc[0][1]);
    acc[1][0] = wmma16x16x32(af[1].v, bf[0].v, acc[1][0]);
    acc[1][1] = wmma16x16x32(af[1].v, bf[1].v, acc[1][1]);
  }

  // epilogue: NHWC packed v8h stores (8 contiguous channels per lane)
#pragma unroll
  for (int mi = 0; mi < 2; ++mi)
#pragma unroll
    for (int ni = 0; ni < 2; ++ni) {
      int kpb = (2 * wm + mi) * 16 + hi * 8;
      int px  = px0 + (2 * wn + ni) * 16 + lr;
      v8h pk;
#pragma unroll
      for (int r = 0; r < 8; ++r) pk[r] = (h16)acc[mi][ni][r];
      *(v8h*)&corr[((size_t)(b * 1024 + px)) * 64 + kpb] = pk;
    }
}

// ===========================================================================
// Kernel 2: pooled mean + 2-layer channel attention (per batch), NHWC corr.
// ===========================================================================
__global__ __launch_bounds__(256) void att_kernel(
    const h16* __restrict__ corr,
    const float* __restrict__ w1, const float* __restrict__ b1,
    const float* __restrict__ w2, const float* __restrict__ b2,
    float* __restrict__ att)
{
  const int b = blockIdx.x, tid = threadIdx.x;
  __shared__ float red[256];
  __shared__ float pooled[64], h1[64];

  const int kp = tid & 63, g = tid >> 6;     // 4 groups of 256 pixels
  float s = 0.f;
  const h16* base = corr + ((size_t)b * 1024 + g * 256) * 64 + kp;
  for (int j = 0; j < 256; ++j) s += (float)base[(size_t)j * 64];
  red[tid] = s;
  __syncthreads();
  if (tid < 64)
    pooled[tid] = (red[tid] + red[tid + 64] + red[tid + 128] + red[tid + 192]) * (1.f / 1024.f);
  __syncthreads();
  if (tid < 64) {
    float a = b1[tid];
    for (int i = 0; i < 64; ++i) a += pooled[i] * w1[tid * 64 + i];
    h1[tid] = fmaxf(a, 0.f);
  }
  __syncthreads();
  if (tid < 64) {
    float a = b2[tid];
    for (int i = 0; i < 64; ++i) a += h1[i] * w2[tid * 64 + i];
    att[b * 64 + tid] = 1.f / (1.f + __expf(-a));
  }
}

// ===========================================================================
// Kernel 3: in-place  corr *= att  (NHWC: chan = i & 63)
// ===========================================================================
__global__ __launch_bounds__(256) void scale_att(h16* __restrict__ corr,
                                                 const float* __restrict__ att)
{
  size_t i = (size_t)blockIdx.x * 256 + threadIdx.x;
  int b = (int)(i >> 16);
  int k = (int)(i & 63);
  corr[i] = (h16)((float)corr[i] * att[b * 64 + k]);
}

// ===========================================================================
// Kernel 4: weight repack f32 OIHW -> f16 [tap][kchunk][oc][ci32] (OC=256)
// dst flat index i == (((t*KC + kc)*256 + oc)*32 + ci)
// ===========================================================================
__global__ __launch_bounds__(256) void pack_w(const float* __restrict__ src,
                                              h16* __restrict__ dst, int CIN)
{
  const int KC = CIN >> 5;
  const int ksh = (CIN == 64) ? 1 : 3;
  int i = blockIdx.x * 256 + threadIdx.x;
  int ci = i & 31;
  int oc = (i >> 5) & 255;
  int rest = i >> 13;
  int kc = rest & (KC - 1);
  int t  = rest >> ksh;
  int cin = kc * 32 + ci;
  dst[i] = (h16)src[((size_t)(oc * CIN + cin)) * 9 + t];
}

// ===========================================================================
// Kernel 5: conv3x3 SAME + folded BN + ReLU, implicit GEMM via WMMA.
//   in : NHWC f16 (128, 32, 32, CIN)     out: NHWC f16 (128, 32, 32, 256)
//   Block: 128 oc x 64 px (2 output rows). 8 waves, each 2x2 16x16 tiles.
//   lB: input halo patch [4 rows][32 x][32 ci] = straight b128 copy (NHWC).
//   lA: 3 weight taps [128 oc][32 ci] staged by the Tensor Data Mover.
// ===========================================================================
__global__ __launch_bounds__(256) void conv3x3_wmma(
    const h16* __restrict__ in, const h16* __restrict__ wp,
    const float* __restrict__ cb,
    const float* __restrict__ bg, const float* __restrict__ bb,
    const float* __restrict__ bm, const float* __restrict__ bv,
    h16* __restrict__ out, int CIN)
{
  const int KC = CIN >> 5;
  const int b      = blockIdx.z;
  const int ocBase = blockIdx.y * 128;
  const int y0     = blockIdx.x * 2;
  const int tid = threadIdx.x;
  const int wv = tid >> 5, lane = tid & 31, lr = lane & 15, hi = lane >> 4;
  const int waveM = (wv & 3) * 2;   // M tiles {waveM, waveM+1} of 8
  const int waveN = (wv >> 2) * 2;  // N tiles {waveN, waveN+1} of 4
  const int k0off = hi ? 8 : 0;

  __shared__ __align__(16) h16 lA[3 * 128 * 32]; // [tx][oc][ci]   24 KB
  __shared__ __align__(16) h16 lB[4 * 32 * 32];  // [row*32+x][ci]  8 KB

  v8f acc[2][2] = {};
  const v16h zfrag = {};

  for (int kc = 0; kc < KC; ++kc) {
    __syncthreads(); // previous iteration's LDS reads complete
    // ---- stage input patch: rows y0-1..y0+2, 32 ci, 32 x (zeros OOB) ----
    {
      int p    = tid >> 1;            // 0..127 : (row, x)
      int half = tid & 1;             // 16-chan half
      int yy   = y0 - 1 + (p >> 5);
      int x    = p & 31;
      h16* dst = &lB[p * 32 + half * 16];
      if (yy >= 0 && yy < 32) {
        const h16* src = in + ((size_t)(b * 1024 + yy * 32 + x)) * CIN + kc * 32 + half * 16;
        *(v8h*)dst       = *(const v8h*)src;
        *(v8h*)(dst + 8) = *(const v8h*)(src + 8);
      } else {
        const v8h z = {};
        *(v8h*)dst       = z;
        *(v8h*)(dst + 8) = z;
      }
    }
    for (int ty = 0; ty < 3; ++ty) {
      if (ty) __syncthreads(); // previous taps' lA reads complete
      // ---- TDM: DMA 3 weight taps (2D tile: 4096 f16 x 3, stride KC*8192) ----
      if (wv == 0) {
        const h16* gsrc = wp + ((size_t)((ty * 3 + 0) * KC + kc) * 256 + ocBase) * 32;
        unsigned long long ga = (unsigned long long)(const void*)gsrc;
        // generic pointer to LDS carries the LDS byte offset in its low 32 bits
        unsigned lds_a = (unsigned)(unsigned long long)(const void*)&lA[0];
        u32x4 g0;
        g0[0] = 1u;                                        // count=1, user D#
        g0[1] = lds_a;                                     // lds_addr
        g0[2] = (unsigned)(ga & 0xFFFFFFFFull);            // global_addr[31:0]
        g0[3] = (unsigned)((ga >> 32) & 0x1FFFFFFu) | (2u << 30); // addr[56:32] | type=2
        i32x8 g1;
        g1[0] = (int)(1u << 16);                           // data_size=1 (2 bytes)
        g1[1] = (int)(4096u << 16);                        // tensor_dim0 = 4096 (lo16)
        g1[2] = (int)(3u << 16);                           // dim0 hi=0 | tensor_dim1 = 3 (lo16)
        g1[3] = (int)(4096u << 16);                        // dim1 hi=0 | tile_dim0 = 4096
        g1[4] = 3;                                         // tile_dim1 = 3, tile_dim2 = 0
        g1[5] = (int)(KC * 8192);                          // tensor_dim0_stride (elements)
        g1[6] = 0;
        g1[7] = 0;
        const i32x4 gz4 = {};
        const i32x8 gz8 = {};
        __builtin_amdgcn_tensor_load_to_lds(g0, g1, gz4, gz4, gz8, 0);
        __builtin_amdgcn_s_wait_tensorcnt(0);
      }
      __syncthreads();
#pragma unroll
      for (int tx = 0; tx < 3; ++tx) {
        FragAB af[2], bf[2];
#pragma unroll
        for (int mi = 0; mi < 2; ++mi) {
          int ocl = (waveM + mi) * 16 + lr;
          const h16* ap = &lA[tx * 4096 + ocl * 32];
          af[mi].h[0] = *(const v8h*)(ap + k0off);
          af[mi].h[1] = *(const v8h*)(ap + 16 + k0off);
        }
#pragma unroll
        for (int ni = 0; ni < 2; ++ni) {
          int p  = (waveN + ni) * 16 + lr;
          int rr = (p >> 5) + ty;
          int xi = (p & 31) + tx - 1;
          if (xi >= 0 && xi < 32) {
            const h16* bp = &lB[(rr * 32 + xi) * 32];
            bf[ni].h[0] = *(const v8h*)(bp + k0off);
            bf[ni].h[1] = *(const v8h*)(bp + 16 + k0off);
          } else {
            bf[ni].v = zfrag;
          }
        }
        acc[0][0] = wmma16x16x32(af[0].v, bf[0].v, acc[0][0]);
        acc[0][1] = wmma16x16x32(af[0].v, bf[1].v, acc[0][1]);
        acc[1][0] = wmma16x16x32(af[1].v, bf[0].v, acc[1][0]);
        acc[1][1] = wmma16x16x32(af[1].v, bf[1].v, acc[1][1]);
      }
    }
  }

  // ---- epilogue: folded conv-bias + BN + ReLU, NHWC packed b128 stores ----
#pragma unroll
  for (int mi = 0; mi < 2; ++mi)
#pragma unroll
    for (int ni = 0; ni < 2; ++ni) {
      int oc0 = ocBase + (waveM + mi) * 16 + hi * 8;
      int p = (waveN + ni) * 16 + lr;
      int y = y0 + (p >> 5), x = p & 31;
      v8h pk;
#pragma unroll
      for (int r = 0; r < 8; ++r) {
        int oc = oc0 + r;
        float sc   = bg[oc] * rsqrtf(bv[oc] + 1e-5f);
        float bias = cb[oc] * sc + bb[oc] - bm[oc] * sc;
        pk[r] = (h16)fmaxf(acc[mi][ni][r] * sc + bias, 0.f);
      }
      *(v8h*)&out[((size_t)(b * 1024 + y * 32 + x)) * 256 + oc0] = pk;
    }
}

// ===========================================================================
// Kernel 6: small heads (256 -> OCS), NHWC feat, f32 VALU.
//   mode 0: cls = 0.1*(conv+b)       mode 1: box = exp(adjust*(conv+b)+bias_p)
// ===========================================================================
__global__ __launch_bounds__(256) void head_conv(
    const h16* __restrict__ feat, const float* __restrict__ w,
    const float* __restrict__ cb, int OCS, int mode,
    const float* __restrict__ adjust, const float* __restrict__ bias_p,
    float* __restrict__ out)
{
  int idx = blockIdx.x * 256 + threadIdx.x;
  int sh  = (OCS == 4) ? 12 : 10;
  int b   = idx >> sh;
  int oc  = (idx >> 10) & (OCS - 1);   // uniform within a 256-thread block
  int p   = idx & 1023;
  int y = p >> 5, x = p & 31;

  float acc = 0.f;
#pragma unroll
  for (int ty = 0; ty < 3; ++ty) {
    int yy = y + ty - 1;
    if (yy < 0 || yy > 31) continue;
#pragma unroll
    for (int tx = 0; tx < 3; ++tx) {
      int xx = x + tx - 1;
      if (xx < 0 || xx > 31) continue;
      const v8h* pv = (const v8h*)(feat + ((size_t)(b * 1024 + yy * 32 + xx)) * 256);
      const float* wt = w + (size_t)oc * 256 * 9 + ty * 3 + tx;   // + cin*9
      for (int c8 = 0; c8 < 32; ++c8) {
        v8h cv = pv[c8];
#pragma unroll
        for (int j = 0; j < 8; ++j)
          acc += (float)cv[j] * wt[(size_t)(c8 * 8 + j) * 9];
      }
    }
  }
  acc += cb[oc];
  if (mode == 0) {
    out[(size_t)b * 1024 + p] = 0.1f * acc;
  } else {
    out[((size_t)(b * 4 + oc)) * 1024 + p] = __expf(adjust[0] * acc + bias_p[oc]);
  }
}

// ===========================================================================
// Host-side orchestration
// ===========================================================================
extern "C" void kernel_launch(void* const* d_in, const int* in_sizes, int n_in,
                              void* d_out, int out_size, void* d_ws, size_t ws_size,
                              hipStream_t stream) {
  (void)in_sizes; (void)n_in; (void)out_size; (void)ws_size;

  const float* search = (const float*)d_in[0];
  const float* kern   = (const float*)d_in[1];
  const float* bnz_g = (const float*)d_in[2],  *bnz_b = (const float*)d_in[3];
  const float* bnz_m = (const float*)d_in[4],  *bnz_v = (const float*)d_in[5];
  const float* bnx_g = (const float*)d_in[6],  *bnx_b = (const float*)d_in[7];
  const float* bnx_m = (const float*)d_in[8],  *bnx_v = (const float*)d_in[9];
  const float* ca_w1 = (const float*)d_in[10], *ca_b1 = (const float*)d_in[11];
  const float* ca_w2 = (const float*)d_in[12], *ca_b2 = (const float*)d_in[13];
  const float* cls_w0 = (const float*)d_in[14], *cls_b0 = (const float*)d_in[15];
  const float* cls_w  = (const float*)d_in[16], *cls_b  = (const float*)d_in[17];
  const float* cls_bn_g = (const float*)d_in[18], *cls_bn_b = (const float*)d_in[19];
  const float* cls_bn_m = (const float*)d_in[20], *cls_bn_v = (const float*)d_in[21];
  const float* reg_w0 = (const float*)d_in[22], *reg_b0 = (const float*)d_in[23];
  const float* reg_w  = (const float*)d_in[24], *reg_b  = (const float*)d_in[25];
  const float* reg_bn_g = (const float*)d_in[26], *reg_bn_b = (const float*)d_in[27];
  const float* reg_bn_m = (const float*)d_in[28], *reg_bn_v = (const float*)d_in[29];
  const float* bbox_w = (const float*)d_in[30], *bbox_b = (const float*)d_in[31];
  const float* clsp_w = (const float*)d_in[32], *clsp_b = (const float*)d_in[33];
  const float* adjust = (const float*)d_in[34], *bias_p = (const float*)d_in[35];
  float* out = (float*)d_out;

  // workspace carve-up
  char* ws = (char*)d_ws;
  size_t off = 0;
  h16* corr = (h16*)(ws + off); off += (size_t)128 * 64 * 1024 * 2;     // 16 MB
  h16* actA = (h16*)(ws + off); off += (size_t)128 * 256 * 1024 * 2;    // 64 MB
  h16* actB = (h16*)(ws + off); off += (size_t)128 * 256 * 1024 * 2;    // 64 MB
  float* att = (float*)(ws + off); off += (size_t)128 * 64 * 4;
  h16* wpk = (h16*)(ws + off);
  const size_t SZ_S = (size_t)256 * 64 * 9;    // small layer (CIN=64)
  const size_t SZ_L = (size_t)256 * 256 * 9;   // large layer (CIN=256)
  h16* p_cls0 = wpk;
  h16* p_cls1 = p_cls0 + SZ_S;
  h16* p_cls2 = p_cls1 + SZ_L;
  h16* p_cls3 = p_cls2 + SZ_L;
  h16* p_reg0 = p_cls3 + SZ_L;
  h16* p_reg1 = p_reg0 + SZ_S;
  h16* p_reg2 = p_reg1 + SZ_L;
  h16* p_reg3 = p_reg2 + SZ_L;

  // ---- weight repack (L2-hot for all conv layers) ----
  pack_w<<<(int)(SZ_S / 256), 256, 0, stream>>>(cls_w0, p_cls0, 64);
  pack_w<<<(int)(SZ_L / 256), 256, 0, stream>>>(cls_w + 0 * SZ_L, p_cls1, 256);
  pack_w<<<(int)(SZ_L / 256), 256, 0, stream>>>(cls_w + 1 * SZ_L, p_cls2, 256);
  pack_w<<<(int)(SZ_L / 256), 256, 0, stream>>>(cls_w + 2 * SZ_L, p_cls3, 256);
  pack_w<<<(int)(SZ_S / 256), 256, 0, stream>>>(reg_w0, p_reg0, 64);
  pack_w<<<(int)(SZ_L / 256), 256, 0, stream>>>(reg_w + 0 * SZ_L, p_reg1, 256);
  pack_w<<<(int)(SZ_L / 256), 256, 0, stream>>>(reg_w + 1 * SZ_L, p_reg2, 256);
  pack_w<<<(int)(SZ_L / 256), 256, 0, stream>>>(reg_w + 2 * SZ_L, p_reg3, 256);

  // ---- BN + correlation (WMMA) ----
  corr_wmma<<<dim3(8, 128), 256, 0, stream>>>(search, kern,
      bnz_g, bnz_b, bnz_m, bnz_v, bnx_g, bnx_b, bnx_m, bnx_v, corr);

  // ---- channel attention + apply ----
  att_kernel<<<128, 256, 0, stream>>>(corr, ca_w1, ca_b1, ca_w2, ca_b2, att);
  scale_att<<<32768, 256, 0, stream>>>(corr, att);

  const dim3 cgrid(16, 2, 128);
  // ---- cls tower ----
  conv3x3_wmma<<<cgrid, 256, 0, stream>>>(corr, p_cls0, cls_b0,
      cls_bn_g + 0,   cls_bn_b + 0,   cls_bn_m + 0,   cls_bn_v + 0,   actA, 64);
  conv3x3_wmma<<<cgrid, 256, 0, stream>>>(actA, p_cls1, cls_b + 0,
      cls_bn_g + 256, cls_bn_b + 256, cls_bn_m + 256, cls_bn_v + 256, actB, 256);
  conv3x3_wmma<<<cgrid, 256, 0, stream>>>(actB, p_cls2, cls_b + 256,
      cls_bn_g + 512, cls_bn_b + 512, cls_bn_m + 512, cls_bn_v + 512, actA, 256);
  conv3x3_wmma<<<cgrid, 256, 0, stream>>>(actA, p_cls3, cls_b + 512,
      cls_bn_g + 768, cls_bn_b + 768, cls_bn_m + 768, cls_bn_v + 768, actB, 256);
  // cls head -> d_out[524288 ..]
  head_conv<<<512, 256, 0, stream>>>(actB, clsp_w, clsp_b, 1, 0, adjust, bias_p,
                                     out + (size_t)128 * 4 * 1024);

  // ---- reg tower ----
  conv3x3_wmma<<<cgrid, 256, 0, stream>>>(corr, p_reg0, reg_b0,
      reg_bn_g + 0,   reg_bn_b + 0,   reg_bn_m + 0,   reg_bn_v + 0,   actA, 64);
  conv3x3_wmma<<<cgrid, 256, 0, stream>>>(actA, p_reg1, reg_b + 0,
      reg_bn_g + 256, reg_bn_b + 256, reg_bn_m + 256, reg_bn_v + 256, actB, 256);
  conv3x3_wmma<<<cgrid, 256, 0, stream>>>(actB, p_reg2, reg_b + 256,
      reg_bn_g + 512, reg_bn_b + 512, reg_bn_m + 512, reg_bn_v + 512, actA, 256);
  conv3x3_wmma<<<cgrid, 256, 0, stream>>>(actA, p_reg3, reg_b + 512,
      reg_bn_g + 768, reg_bn_b + 768, reg_bn_m + 768, reg_bn_v + 768, actB, 256);
  // box head -> d_out[0 ..]
  head_conv<<<2048, 256, 0, stream>>>(actB, bbox_w, bbox_b, 4, 1, adjust, bias_p,
                                      out);
}
